// transformer_block_33492154974633
// MI455X (gfx1250) — matrix operand order repaired and checked by reading, not compiled, not run on hardware
//
#include <hip/hip_runtime.h>
#include <hip/hip_bf16.h>
#include <math.h>

#define D_MODEL 1024
#define N_HEADS 16
#define D_HEAD  64
#define D_FF    4096
#define BATCH   2
#define SEQ     2048
#define BT      (BATCH * SEQ)
#define EPSF    1e-5f
#define THETAF  10000.0f

typedef __attribute__((ext_vector_type(16))) __bf16 v16bf;
typedef __attribute__((ext_vector_type(8)))  float  v8f;
typedef __attribute__((ext_vector_type(8)))  unsigned short su8;

union Frag {
    v16bf v;
    su8 h[2];
    unsigned short u[16];
};

__device__ inline unsigned short f2bf(float f) {
    union { float f; unsigned u; } c; c.f = f;
    unsigned r = c.u + 0x7FFFu + ((c.u >> 16) & 1u);
    return (unsigned short)(r >> 16);
}
__device__ inline float bf2f(unsigned short h) {
    union { unsigned u; float f; } c; c.u = ((unsigned)h) << 16;
    return c.f;
}

__device__ inline v8f wmma_bf16(const Frag& a, const Frag& b, v8f c) {
    return __builtin_amdgcn_wmma_f32_16x16x32_bf16(
        /*neg_a=*/false, a.v, /*neg_b=*/false, b.v,
        /*c_mod=*/(short)0, c, /*reuse_a=*/false, /*reuse_b=*/false);
}

// ---------------------------------------------------------------------------
// f32 -> bf16 conversion (weights)
// ---------------------------------------------------------------------------
__global__ void cvt_bf16_kernel(const float* __restrict__ in,
                                unsigned short* __restrict__ out, int n) {
    int i = blockIdx.x * blockDim.x + threadIdx.x;
    if (i < n) out[i] = f2bf(in[i]);
}

// ---------------------------------------------------------------------------
// RMSNorm: out_bf16[row,:] = x[row,:] * g[:] * rsqrt(mean(x^2)+eps)
// one block (256 threads) per row of D_MODEL
// ---------------------------------------------------------------------------
__global__ void __launch_bounds__(256)
rmsnorm_bf16_kernel(const float* __restrict__ x, const float* __restrict__ g,
                    unsigned short* __restrict__ out) {
    __shared__ float red[8];
    const int row = blockIdx.x;
    const float* xr = x + (size_t)row * D_MODEL;
    float ss = 0.f;
    for (int i = threadIdx.x; i < D_MODEL; i += 256) {
        float v = xr[i]; ss += v * v;
    }
    for (int off = 16; off > 0; off >>= 1) ss += __shfl_down(ss, off, 32);
    if ((threadIdx.x & 31) == 0) red[threadIdx.x >> 5] = ss;
    __syncthreads();
    float tot = 0.f;
    #pragma unroll
    for (int i = 0; i < 8; ++i) tot += red[i];
    const float rms = rsqrtf(tot / (float)D_MODEL + EPSF);
    unsigned short* orow = out + (size_t)row * D_MODEL;
    for (int i = threadIdx.x; i < D_MODEL; i += 256)
        orow[i] = f2bf(xr[i] * g[i] * rms);
}

// ---------------------------------------------------------------------------
// WMMA NT-GEMM: out[m,n] = sum_k A[m,k] * W[n,k]
// A: [M,K] bf16 row-major, W: [N,K] bf16 row-major.
// Block = 256 threads = 8 waves; wave tile 32x64 (2 A-frags x 4 B-frags ->
// 8 WMMA per 32-wide K step, B-fragments reused across both A rows).
// Block tile 64x256.  grid.x = N/256, grid.y = M/64.
// ---------------------------------------------------------------------------
__global__ void __launch_bounds__(256)
gemm_bf16_nt_kernel(const unsigned short* __restrict__ A,
                    const unsigned short* __restrict__ W,
                    float* __restrict__ outF,
                    unsigned short* __restrict__ outH,
                    int M, int N, int K) {
    const int lane = threadIdx.x & 31;
    const int wave = threadIdx.x >> 5;
    const int lr = lane & 15;
    const int lh = lane >> 4;
    const int m0 = blockIdx.y * 64 + (wave >> 2) * 32;
    const int n0 = blockIdx.x * 256 + (wave & 3) * 64;

    v8f acc[2][4] = {};

    const size_t aRow0 = (size_t)(m0 +  0 + lr) * K;
    const size_t aRow1 = (size_t)(m0 + 16 + lr) * K;
    const size_t bRow0 = (size_t)(n0 +  0 + lr) * K;
    const size_t bRow1 = (size_t)(n0 + 16 + lr) * K;
    const size_t bRow2 = (size_t)(n0 + 32 + lr) * K;
    const size_t bRow3 = (size_t)(n0 + 48 + lr) * K;

    for (int k0 = 0; k0 < K; k0 += 32) {
        const int ka = k0 + (lh << 4);        // A: hi lanes take K +16
        Frag a0, a1;
        a0.h[0] = *(const su8*)(A + aRow0 + ka);
        a0.h[1] = *(const su8*)(A + aRow0 + ka + 8);
        a1.h[0] = *(const su8*)(A + aRow1 + ka);
        a1.h[1] = *(const su8*)(A + aRow1 + ka + 8);

        const int kb = k0 + (lh << 4);        // B: hi lanes take K +16
        Frag b0, b1, b2, b3;
        b0.h[0] = *(const su8*)(W + bRow0 + kb);
        b0.h[1] = *(const su8*)(W + bRow0 + kb + 8);
        b1.h[0] = *(const su8*)(W + bRow1 + kb);
        b1.h[1] = *(const su8*)(W + bRow1 + kb + 8);
        b2.h[0] = *(const su8*)(W + bRow2 + kb);
        b2.h[1] = *(const su8*)(W + bRow2 + kb + 8);
        b3.h[0] = *(const su8*)(W + bRow3 + kb);
        b3.h[1] = *(const su8*)(W + bRow3 + kb + 8);

        acc[0][0] = wmma_bf16(a0, b0, acc[0][0]);
        acc[1][0] = wmma_bf16(a1, b0, acc[1][0]);
        acc[0][1] = wmma_bf16(a0, b1, acc[0][1]);
        acc[1][1] = wmma_bf16(a1, b1, acc[1][1]);
        acc[0][2] = wmma_bf16(a0, b2, acc[0][2]);
        acc[1][2] = wmma_bf16(a1, b2, acc[1][2]);
        acc[0][3] = wmma_bf16(a0, b3, acc[0][3]);
        acc[1][3] = wmma_bf16(a1, b3, acc[1][3]);
    }

    #pragma unroll
    for (int mi = 0; mi < 2; ++mi) {
        #pragma unroll
        for (int j = 0; j < 4; ++j) {
            #pragma unroll
            for (int r = 0; r < 8; ++r) {
                const int row = m0 + mi * 16 + lh * 8 + r;
                const int col = n0 + j * 16 + lr;
                const float vv = acc[mi][j][r];
                if (outF) outF[(size_t)row * N + col] = vv;
                else      outH[(size_t)row * N + col] = f2bf(vv);
            }
        }
    }
}

// ---------------------------------------------------------------------------
// RoPE + QKV split: qkv f32 [BT, 3C] -> q,k,v bf16 [B,H,T,dh]
// ---------------------------------------------------------------------------
__global__ void rope_split_kernel(const float* __restrict__ qkv,
                                  unsigned short* __restrict__ qo,
                                  unsigned short* __restrict__ ko,
                                  unsigned short* __restrict__ vo) {
    const int ROT = D_HEAD / 2;
    int idx = blockIdx.x * blockDim.x + threadIdx.x;
    if (idx >= BT * N_HEADS * ROT) return;
    const int i   = idx % ROT;
    const int h   = (idx / ROT) % N_HEADS;
    const int tok = idx / (ROT * N_HEADS);       // b*T + t
    const int t   = tok % SEQ;
    const int b   = tok / SEQ;

    const float inv = __expf(-(2.0f * (float)i) / (float)D_HEAD * __logf(THETAF));
    const float th  = (float)t * inv;
    const float c = __cosf(th), s = __sinf(th);

    const size_t bi = (size_t)tok * (3 * D_MODEL) + (size_t)h * D_HEAD;
    const float q1 = qkv[bi + 2 * i],                 q2 = qkv[bi + 2 * i + 1];
    const float k1 = qkv[bi + D_MODEL + 2 * i],       k2 = qkv[bi + D_MODEL + 2 * i + 1];
    const float v1 = qkv[bi + 2 * D_MODEL + 2 * i],   v2 = qkv[bi + 2 * D_MODEL + 2 * i + 1];

    const size_t bo = ((size_t)(b * N_HEADS + h) * SEQ + t) * D_HEAD;
    qo[bo + 2 * i]     = f2bf(q1 * c - q2 * s);
    qo[bo + 2 * i + 1] = f2bf(q1 * s + q2 * c);
    ko[bo + 2 * i]     = f2bf(k1 * c - k2 * s);
    ko[bo + 2 * i + 1] = f2bf(k1 * s + k2 * c);
    vo[bo + 2 * i]     = f2bf(v1);
    vo[bo + 2 * i + 1] = f2bf(v2);
}

// ---------------------------------------------------------------------------
// Causal flash attention, WMMA. One wave per (b, h, 16-row q-tile).
// q,k,v bf16 [B,H,T,dh]; o bf16 [B,T,H*dh].
// V tiles staged to LDS with gfx1250 async-to-LDS (ASYNCcnt), overlapping
// the S = Q K^T WMMAs + softmax; P.V B-fragments then gather from LDS.
// ---------------------------------------------------------------------------
__global__ void __launch_bounds__(32)
attn_fwd_kernel(const unsigned short* __restrict__ q,
                const unsigned short* __restrict__ k,
                const unsigned short* __restrict__ v,
                unsigned short* __restrict__ o) {
    __shared__ float          sS[16][33];
    __shared__ unsigned short sP[16][32];
    __shared__ unsigned short sV[32][64];   // one 32-kv x 64-d V tile
    __shared__ float sM[16], sL[16], sC[16];

    const int lane = threadIdx.x;
    const int lr = lane & 15, lh = lane >> 4;
    const int qt = blockIdx.x, h = blockIdx.y, b = blockIdx.z;
    const size_t head_base = (size_t)(b * N_HEADS + h) * SEQ * D_HEAD;
    const float scale = rsqrtf((float)D_HEAD);

    // Q A-fragments (two 16x32 chunks over dh=64), loaded once.
    Frag aQ0, aQ1;
    {
        const size_t qrow = head_base + (size_t)(qt * 16 + lr) * D_HEAD;
        const int ka = lh << 4;
        aQ0.h[0] = *(const su8*)(q + qrow + ka);
        aQ0.h[1] = *(const su8*)(q + qrow + ka + 8);
        aQ1.h[0] = *(const su8*)(q + qrow + 32 + ka);
        aQ1.h[1] = *(const su8*)(q + qrow + 32 + ka + 8);
    }

    if (lane < 16) { sM[lane] = -1e30f; sL[lane] = 0.f; }
    __syncthreads();

    v8f oacc[4] = {};
    const unsigned ldsV = (unsigned)(size_t)(&sV[lane][0]);  // per-lane V row

    const int ktMax = (qt * 16 + 15) >> 5;
    for (int kt = 0; kt <= ktMax; ++kt) {
        // ---- kick off async V-tile copy (32 rows x 128B); lane owns row ----
        {
            const unsigned long long ga =
                (unsigned long long)(size_t)(v + head_base +
                                             (size_t)(kt * 32 + lane) * D_HEAD);
            #pragma unroll
            for (int c = 0; c < 8; ++c) {
                asm volatile("global_load_async_to_lds_b128 %0, %1, off"
                             :: "v"(ldsV + c * 16), "v"(ga + c * 16)
                             : "memory");
            }
        }

        // ---- S = Q K^T (16 q-rows x 32 kv-cols), 4 WMMA ----
        v8f s0 = {}, s1 = {};
        {
            const size_t kb0 = head_base + (size_t)(kt * 32 +      lr) * D_HEAD;
            const size_t kb1 = head_base + (size_t)(kt * 32 + 16 + lr) * D_HEAD;
            const int kk = lh << 4;
            Frag bf;
            bf.h[0] = *(const su8*)(k + kb0 + kk);
            bf.h[1] = *(const su8*)(k + kb0 + kk + 8);
            s0 = wmma_bf16(aQ0, bf, s0);
            bf.h[0] = *(const su8*)(k + kb0 + 32 + kk);
            bf.h[1] = *(const su8*)(k + kb0 + 32 + kk + 8);
            s0 = wmma_bf16(aQ1, bf, s0);
            bf.h[0] = *(const su8*)(k + kb1 + kk);
            bf.h[1] = *(const su8*)(k + kb1 + kk + 8);
            s1 = wmma_bf16(aQ0, bf, s1);
            bf.h[0] = *(const su8*)(k + kb1 + 32 + kk);
            bf.h[1] = *(const su8*)(k + kb1 + 32 + kk + 8);
            s1 = wmma_bf16(aQ1, bf, s1);
        }

        // ---- scale + causal mask, spill S tile to LDS ----
        #pragma unroll
        for (int r = 0; r < 8; ++r) {
            const int row = lh * 8 + r;
            const int qg = qt * 16 + row;
            const int kv0 = kt * 32 + lr;
            const int kv1 = kt * 32 + 16 + lr;
            sS[row][lr]      = (kv0 <= qg) ? s0[r] * scale : -1e30f;
            sS[row][16 + lr] = (kv1 <= qg) ? s1[r] * scale : -1e30f;
        }
        __syncthreads();

        // ---- online softmax per row (lanes 0..15 own one row each) ----
        if (lane < 16) {
            const float m_old = sM[lane];
            float tm = -1e30f;
            for (int c = 0; c < 32; ++c) tm = fmaxf(tm, sS[lane][c]);
            const float m_new = fmaxf(m_old, tm);
            const float corr = __expf(m_old - m_new);
            float ls = 0.f;
            for (int c = 0; c < 32; ++c) {
                const float p = __expf(sS[lane][c] - m_new);
                ls += p;
                sP[lane][c] = f2bf(p);
            }
            sM[lane] = m_new;
            sL[lane] = sL[lane] * corr + ls;
            sC[lane] = corr;
        }
        __syncthreads();

        // ---- rescale accumulators ----
        #pragma unroll
        for (int r = 0; r < 8; ++r) {
            const float f = sC[lh * 8 + r];
            oacc[0][r] *= f; oacc[1][r] *= f; oacc[2][r] *= f; oacc[3][r] *= f;
        }

        // ---- P fragment (16x32 bf16 from LDS) ----
        Frag aP;
        #pragma unroll
        for (int e = 0; e < 8; ++e) aP.u[e]     = sP[lr][lh * 8 + e];
        #pragma unroll
        for (int e = 0; e < 8; ++e) aP.u[8 + e] = sP[lr][16 + lh * 8 + e];

        // ---- wait for async V tile, then O += P V (4 WMMA) ----
        asm volatile("s_wait_asynccnt 0" ::: "memory");
        __syncthreads();

        #pragma unroll
        for (int j = 0; j < 4; ++j) {
            Frag bV;
            #pragma unroll
            for (int e = 0; e < 16; ++e)
                bV.u[e] = sV[(e & 8) * 2 + lh * 8 + (e & 7)][j * 16 + lr];
            oacc[j] = wmma_bf16(aP, bV, oacc[j]);
        }
        __syncthreads();
    }

    // ---- normalize and write [B, T, H*dh] ----
    #pragma unroll
    for (int j = 0; j < 4; ++j) {
        #pragma unroll
        for (int r = 0; r < 8; ++r) {
            const int row = lh * 8 + r;
            const float li = sL[row];
            const int qg = qt * 16 + row;
            const size_t off = ((size_t)(b * SEQ) + qg) * D_MODEL
                             + (size_t)h * D_HEAD + j * 16 + lr;
            o[off] = f2bf(oacc[j][r] / li);
        }
    }
}

// ---------------------------------------------------------------------------
// SwiGLU gate: g = silu(a) * b   (bf16 in/out)
// ---------------------------------------------------------------------------
__global__ void swiglu_kernel(const unsigned short* __restrict__ a,
                              const unsigned short* __restrict__ b,
                              unsigned short* __restrict__ g, int n) {
    int i = blockIdx.x * blockDim.x + threadIdx.x;
    if (i >= n) return;
    const float av = bf2f(a[i]);
    const float bv = bf2f(b[i]);
    const float s = av / (1.f + __expf(-av));
    g[i] = f2bf(s * bv);
}

// ---------------------------------------------------------------------------
// out = a + b (f32)
// ---------------------------------------------------------------------------
__global__ void add_f32_kernel(const float* __restrict__ a,
                               const float* __restrict__ b,
                               float* __restrict__ out, int n) {
    int i = blockIdx.x * blockDim.x + threadIdx.x;
    if (i < n) out[i] = a[i] + b[i];
}

// ---------------------------------------------------------------------------
// Host-side orchestration
// ---------------------------------------------------------------------------
extern "C" void kernel_launch(void* const* d_in, const int* in_sizes, int n_in,
                              void* d_out, int out_size, void* d_ws, size_t ws_size,
                              hipStream_t stream) {
    const float* x      = (const float*)d_in[0];
    const float* w_qkv  = (const float*)d_in[1];
    const float* w_proj = (const float*)d_in[2];
    const float* g1     = (const float*)d_in[3];
    const float* g2     = (const float*)d_in[4];
    const float* w1     = (const float*)d_in[5];
    const float* w2     = (const float*)d_in[6];
    const float* w3     = (const float*)d_in[7];
    float* out = (float*)d_out;

    char* ws = (char*)d_ws;
    size_t off = 0;
    auto alloc = [&](size_t bytes) -> void* {
        off = (off + 255) & ~(size_t)255;
        void* p = ws + off;
        off += bytes;
        return p;
    };

    unsigned short* wqkv_h  = (unsigned short*)alloc((size_t)3 * D_MODEL * D_MODEL * 2);
    unsigned short* wproj_h = (unsigned short*)alloc((size_t)D_MODEL * D_MODEL * 2);
    unsigned short* w1_h    = (unsigned short*)alloc((size_t)D_FF * D_MODEL * 2);
    unsigned short* w2_h    = (unsigned short*)alloc((size_t)D_MODEL * D_FF * 2);
    unsigned short* w3_h    = (unsigned short*)alloc((size_t)D_FF * D_MODEL * 2);
    unsigned short* hn      = (unsigned short*)alloc((size_t)BT * D_MODEL * 2);
    float*          qkv_f   = (float*)alloc((size_t)BT * 3 * D_MODEL * 4);
    unsigned short* qh      = (unsigned short*)alloc((size_t)BT * D_MODEL * 2);
    unsigned short* kh      = (unsigned short*)alloc((size_t)BT * D_MODEL * 2);
    unsigned short* vh      = (unsigned short*)alloc((size_t)BT * D_MODEL * 2);
    unsigned short* oh      = (unsigned short*)alloc((size_t)BT * D_MODEL * 2);
    float*          proj_f  = (float*)alloc((size_t)BT * D_MODEL * 4);
    float*          x1      = (float*)alloc((size_t)BT * D_MODEL * 4);
    unsigned short* h2n     = (unsigned short*)alloc((size_t)BT * D_MODEL * 2);
    unsigned short* a_h     = (unsigned short*)alloc((size_t)BT * D_FF * 2);
    unsigned short* b_h     = (unsigned short*)alloc((size_t)BT * D_FF * 2);
    unsigned short* g_h     = (unsigned short*)alloc((size_t)BT * D_FF * 2);
    float*          ff_f    = (float*)alloc((size_t)BT * D_MODEL * 4);
    (void)ws_size; (void)in_sizes; (void)n_in; (void)out_size;

    auto cvt = [&](const float* src, unsigned short* dst, int n) {
        cvt_bf16_kernel<<<(n + 255) / 256, 256, 0, stream>>>(src, dst, n);
    };
    cvt(w_qkv,  wqkv_h,  3 * D_MODEL * D_MODEL);
    cvt(w_proj, wproj_h, D_MODEL * D_MODEL);
    cvt(w1,     w1_h,    D_FF * D_MODEL);
    cvt(w2,     w2_h,    D_MODEL * D_FF);
    cvt(w3,     w3_h,    D_FF * D_MODEL);

    // h = rmsnorm(x, g1) -> bf16
    rmsnorm_bf16_kernel<<<BT, 256, 0, stream>>>(x, g1, hn);

    // qkv = h @ w_qkv.T  (f32 out)
    gemm_bf16_nt_kernel<<<dim3(3 * D_MODEL / 256, BT / 64), 256, 0, stream>>>(
        hn, wqkv_h, qkv_f, nullptr, BT, 3 * D_MODEL, D_MODEL);

    // rope + split -> q,k,v bf16 [B,H,T,dh]
    {
        const int n = BT * N_HEADS * (D_HEAD / 2);
        rope_split_kernel<<<(n + 255) / 256, 256, 0, stream>>>(qkv_f, qh, kh, vh);
    }

    // attention -> o bf16 [BT, C]
    attn_fwd_kernel<<<dim3(SEQ / 16, N_HEADS, BATCH), 32, 0, stream>>>(qh, kh, vh, oh);

    // proj = o @ w_proj.T (f32), x1 = x + proj
    gemm_bf16_nt_kernel<<<dim3(D_MODEL / 256, BT / 64), 256, 0, stream>>>(
        oh, wproj_h, proj_f, nullptr, BT, D_MODEL, D_MODEL);
    add_f32_kernel<<<(BT * D_MODEL + 255) / 256, 256, 0, stream>>>(
        x, proj_f, x1, BT * D_MODEL);

    // h2 = rmsnorm(x1, g2) -> bf16
    rmsnorm_bf16_kernel<<<BT, 256, 0, stream>>>(x1, g2, h2n);

    // a = h2 @ w1.T, b = h2 @ w3.T  (bf16 out)
    gemm_bf16_nt_kernel<<<dim3(D_FF / 256, BT / 64), 256, 0, stream>>>(
        h2n, w1_h, nullptr, a_h, BT, D_FF, D_MODEL);
    gemm_bf16_nt_kernel<<<dim3(D_FF / 256, BT / 64), 256, 0, stream>>>(
        h2n, w3_h, nullptr, b_h, BT, D_FF, D_MODEL);

    // g = silu(a) * b
    swiglu_kernel<<<(BT * D_FF + 255) / 256, 256, 0, stream>>>(
        a_h, b_h, g_h, BT * D_FF);

    // ff = g @ w2.T (f32), out = x1 + ff
    gemm_bf16_nt_kernel<<<dim3(D_MODEL / 256, BT / 64), 256, 0, stream>>>(
        g_h, w2_h, ff_f, nullptr, BT, D_MODEL, D_FF);
    add_f32_kernel<<<(BT * D_MODEL + 255) / 256, 256, 0, stream>>>(
        x1, ff_f, out, BT * D_MODEL);
}
